// UniRouteMoELayer_18150531793245
// MI455X (gfx1250) — matrix operand order repaired and checked by reading, not compiled, not run on hardware
//
#include <hip/hip_runtime.h>
#include <hip/hip_bf16.h>
#include <math.h>

// Problem sizes (fixed by reference)
#define BB 512
#define TT 32
#define HH 768
#define FF 3072
#define EE 3
#define FC 256   // F-chunk staged through LDS

typedef __attribute__((ext_vector_type(16))) __bf16 v16bf;
typedef __attribute__((ext_vector_type(8)))  float v8f;

union BF16Frag {
    v16bf v;
    uint4 q[2];
    unsigned short u[16];
};

__device__ __forceinline__ unsigned short f2bf(float f) {
    unsigned int u = __builtin_bit_cast(unsigned int, f);
    // round-to-nearest-even truncation to bf16
    unsigned int r = u + 0x7FFFu + ((u >> 16) & 1u);
    return (unsigned short)(r >> 16);
}

// ---------------------------------------------------------------------------
// Kernel 1: router.  scores = softmax(mean_t(x[b]) @ Wg^T); argmax -> gate,sel
// ---------------------------------------------------------------------------
__global__ __launch_bounds__(256) void moe_router(const float* __restrict__ x,
                                                  const float* __restrict__ Wg,
                                                  float* __restrict__ gate,
                                                  int* __restrict__ sel) {
    __shared__ float red[EE][256];
    const int b = blockIdx.x;
    const int tid = threadIdx.x;
    const float* xb = x + (size_t)b * TT * HH;

    float p0 = 0.f, p1 = 0.f, p2 = 0.f;
    for (int c = tid; c < HH; c += 256) {
        float s = 0.f;
        #pragma unroll
        for (int t = 0; t < TT; ++t) s += xb[t * HH + c];
        const float m = s * (1.0f / TT);
        p0 += m * Wg[0 * HH + c];
        p1 += m * Wg[1 * HH + c];
        p2 += m * Wg[2 * HH + c];
    }
    red[0][tid] = p0; red[1][tid] = p1; red[2][tid] = p2;
    __syncthreads();
    for (int s = 128; s > 0; s >>= 1) {
        if (tid < s) {
            red[0][tid] += red[0][tid + s];
            red[1][tid] += red[1][tid + s];
            red[2][tid] += red[2][tid + s];
        }
        __syncthreads();
    }
    if (tid == 0) {
        const float s0 = red[0][0], s1 = red[1][0], s2 = red[2][0];
        const float mx = fmaxf(s0, fmaxf(s1, s2));
        const float e0 = __expf(s0 - mx), e1 = __expf(s1 - mx), e2 = __expf(s2 - mx);
        const float inv = 1.0f / (e0 + e1 + e2);
        int id = 0; float best = s0;
        if (s1 > best) { best = s1; id = 1; }
        if (s2 > best) { best = s2; id = 2; }
        gate[b] = (id == 0 ? e0 : (id == 1 ? e1 : e2)) * inv;
        sel[b]  = id;
    }
}

// ---------------------------------------------------------------------------
// Kernel 1b: weight pre-pass.  src (E x R x C f32, row-major) ->
// dst (E x C x R bf16), transposed so the GEMM K-dim is contiguous per
// output column.  Tiled through LDS so both sides stay coalesced.
// ---------------------------------------------------------------------------
__global__ __launch_bounds__(256) void w_transpose_bf16(const float* __restrict__ src,
                                                        unsigned short* __restrict__ dst,
                                                        int R, int C) {
    __shared__ unsigned short tile[32][33];
    const size_t off = (size_t)blockIdx.z * R * C;
    const int c0 = blockIdx.x * 32, r0 = blockIdx.y * 32;
    const int tx = threadIdx.x & 31, ty = threadIdx.x >> 5;  // 32 x 8
    #pragma unroll
    for (int i = 0; i < 32; i += 8)
        tile[ty + i][tx] = f2bf(src[off + (size_t)(r0 + ty + i) * C + c0 + tx]);
    __syncthreads();
    #pragma unroll
    for (int i = 0; i < 32; i += 8)
        dst[off + (size_t)(c0 + ty + i) * R + r0 + tx] = tile[tx][ty + i];
}

// ---------------------------------------------------------------------------
// Kernel 2: fused expert FFN, one batch per block, 8 waves.
//   out = gate * (gelu(x @ W1[e] + b1[e]) @ W2[e] + b2[e])
// Weights pre-transposed bf16: W1t[e][f][k], W2t[e][h][k] -> every B fragment
// is two contiguous 16-byte GLOBAL loads.  Addresses are a single fixed
// global base pointer per operand + integer offsets (unroll-constant parts
// fold into the 24-bit instruction offset; no generic/FLAT pointers).
// ---------------------------------------------------------------------------
__global__ __launch_bounds__(256) void moe_ffn(const float* __restrict__ x,
                                               const unsigned short* __restrict__ W1t,
                                               const float* __restrict__ b1,
                                               const unsigned short* __restrict__ W2t,
                                               const float* __restrict__ b2,
                                               const float* __restrict__ gate,
                                               const int* __restrict__ sel,
                                               float* __restrict__ out) {
    __shared__ unsigned short xs[TT][HH];  // 48 KB: x[b] in bf16
    __shared__ unsigned short hs[TT][FC];  // 16 KB: gelu chunk in bf16

    const int b    = blockIdx.x;
    const int tid  = threadIdx.x;
    const int lane = tid & 31;
    const int wave = tid >> 5;
    const int m16  = lane & 15;   // row/col within 16-wide tile
    const int hseg = lane >> 4;   // which half of the wave

    const int   e = sel[b];
    const float g = gate[b];

    const float* xb = x + (size_t)b * TT * HH;
    const float* b1e = b1 + (size_t)e * FF;
    const float* b2e = b2 + (size_t)e * HH;

    // ---- stage x[b] -> LDS as bf16 (vectorized, coalesced) ----
    {
        unsigned short* xsf = &xs[0][0];
        for (int i = tid * 4; i < TT * HH; i += 256 * 4) {
            const float4 v = *(const float4*)(xb + i);
            xsf[i + 0] = f2bf(v.x);
            xsf[i + 1] = f2bf(v.y);
            xsf[i + 2] = f2bf(v.z);
            xsf[i + 3] = f2bf(v.w);
        }
    }
    __syncthreads();

    // Persistent output accumulators: wave covers rows [mtY*16,+16),
    // 12 column tiles of 16 starting at (wave&3)*192.
    const int mtY = wave >> 2;
    const int ntY = (wave & 3) * 12;
    v8f yacc[12];
    #pragma unroll
    for (int t = 0; t < 12; ++t)
        yacc[t] = (v8f){0.f, 0.f, 0.f, 0.f, 0.f, 0.f, 0.f, 0.f};

    const int mtH = wave & 1;          // phase-1 row tile
    const int ftH = (wave >> 1) * 4;   // phase-1: 4 f-subtiles of 16

    // Fixed global bases (never rewritten -> stay addrspace(1));
    // all loop-variant parts are integer offsets.
    const unsigned short* w1p =
        W1t + (size_t)e * HH * FF + (size_t)(ftH * 16 + m16) * HH + hseg * 16;
    const unsigned short* w2p =
        W2t + (size_t)e * FF * HH + (size_t)((ntY) * 16 + m16) * FF + hseg * 16;

    const unsigned short* arow1 = &xs[mtH * 16 + m16][0];
    const unsigned short* arow2 = &hs[mtY * 16 + m16][0];
    const int kA = hseg * 16;

    int w1off = 0;  // advances by FC*HH per chunk
    int w2off = 0;  // advances by FC per chunk

    for (int fc0 = 0; fc0 < FF; fc0 += FC) {
        // ---------------- phase 1: h = gelu(x @ W1 + b1) ----------------
        v8f hacc[4];
        #pragma unroll
        for (int t = 0; t < 4; ++t)
            hacc[t] = (v8f){0.f, 0.f, 0.f, 0.f, 0.f, 0.f, 0.f, 0.f};

        for (int kb = 0; kb < HH; kb += 32) {
            BF16Frag A;  // row m = mtH*16+m16; one contiguous 16-elem K run
            A.q[0] = *(const uint4*)(arow1 + kb + kA);
            A.q[1] = *(const uint4*)(arow1 + kb + kA + 8);
            #pragma unroll
            for (int t = 0; t < 4; ++t) {
                const uint4* qp = (const uint4*)(w1p + w1off + t * 16 * HH + kb);
                BF16Frag Bf;
                Bf.q[0] = qp[0];
                Bf.q[1] = qp[1];
                hacc[t] = __builtin_amdgcn_wmma_f32_16x16x32_bf16(
                    false, A.v, false, Bf.v, (short)0, hacc[t], false, false);
            }
        }

        __syncthreads();  // previous phase-2 readers of hs are done
        #pragma unroll
        for (int t = 0; t < 4; ++t) {
            const int fcol = (ftH + t) * 16 + m16;  // column within chunk
            const float bias = b1e[fc0 + fcol];
            #pragma unroll
            for (int r = 0; r < 8; ++r) {
                float v = hacc[t][r] + bias;
                v = 0.5f * v * (1.0f + erff(v * 0.70710678118654752f));  // exact gelu
                hs[mtH * 16 + hseg * 8 + r][fcol] = f2bf(v);
            }
        }
        __syncthreads();

        // ---------------- phase 2: y += h @ W2 ----------------
        for (int kb = 0; kb < FC; kb += 32) {
            BF16Frag A;
            A.q[0] = *(const uint4*)(arow2 + kb + kA);
            A.q[1] = *(const uint4*)(arow2 + kb + kA + 8);
            #pragma unroll
            for (int t = 0; t < 12; ++t) {
                const uint4* qp = (const uint4*)(w2p + w2off + t * 16 * FF + kb);
                BF16Frag Bf;
                Bf.q[0] = qp[0];
                Bf.q[1] = qp[1];
                yacc[t] = __builtin_amdgcn_wmma_f32_16x16x32_bf16(
                    false, A.v, false, Bf.v, (short)0, yacc[t], false, false);
            }
        }
        w1off += FC * HH;
        w2off += FC;
        __syncthreads();
    }

    // ---- epilogue: out = gate * (y + b2); non-temporal so the streaming
    // output does not evict the L2-resident bf16 weights. ----
    float* ob = out + (size_t)b * TT * HH;
    #pragma unroll
    for (int t = 0; t < 12; ++t) {
        const int hcol = (ntY + t) * 16 + m16;
        const float bias = b2e[hcol];
        #pragma unroll
        for (int r = 0; r < 8; ++r) {
            const int m = mtY * 16 + hseg * 8 + r;
            __builtin_nontemporal_store(g * (yacc[t][r] + bias),
                                        ob + (size_t)m * HH + hcol);
        }
    }
}

// ---------------------------------------------------------------------------
extern "C" void kernel_launch(void* const* d_in, const int* in_sizes, int n_in,
                              void* d_out, int out_size, void* d_ws, size_t ws_size,
                              hipStream_t stream) {
    const float* x  = (const float*)d_in[0];
    const float* Wg = (const float*)d_in[1];
    const float* W1 = (const float*)d_in[2];
    const float* b1 = (const float*)d_in[3];
    const float* W2 = (const float*)d_in[4];
    const float* b2 = (const float*)d_in[5];
    float* out = (float*)d_out;

    // Workspace layout: gate (512 f32) | sel (512 i32) | pad | W1t | W2t (bf16)
    float* gate = (float*)d_ws;
    int*   sel  = (int*)((char*)d_ws + BB * sizeof(float));
    unsigned short* W1t = (unsigned short*)((char*)d_ws + 8192);
    unsigned short* W2t = W1t + (size_t)EE * HH * FF;

    moe_router<<<BB, 256, 0, stream>>>(x, Wg, gate, sel);
    // W1: E x (HH x FF) -> W1t: E x (FF x HH)
    w_transpose_bf16<<<dim3(FF / 32, HH / 32, EE), 256, 0, stream>>>(W1, W1t, HH, FF);
    // W2: E x (FF x HH) -> W2t: E x (HH x FF)
    w_transpose_bf16<<<dim3(HH / 32, FF / 32, EE), 256, 0, stream>>>(W2, W2t, FF, HH);
    moe_ffn<<<BB, 256, 0, stream>>>(x, W1t, b1, W2t, b2, gate, sel, out);
}
